// ConsciousWorkingMemory_32959579029635
// MI455X (gfx1250) — compile-verified, matrix operand order checked by reading, not run on hardware
//
#include <hip/hip_runtime.h>
#include <cstdint>
#include <cstddef>

typedef __bf16 bf16;
typedef __attribute__((ext_vector_type(16))) __bf16 v16bf;
typedef __attribute__((ext_vector_type(8)))  float  v8f;
typedef __attribute__((ext_vector_type(4)))  unsigned int u32x4;
typedef __attribute__((ext_vector_type(8)))  int          i32x8;
typedef __attribute__((ext_vector_type(4)))  int          i32x4;

#if __has_include(<hip/amd_detail/amd_gfx1250_TDM.h>)
#define TDM_6ARG 1
#else
#define TDM_6ARG 0
#endif

#define SEQ   2048
#define D4    1024
#define BATCH 4
#define MROWS (BATCH * SEQ)   // 8192

// ---------------------------------------------------------------------------
// Tensor Data Mover: 2D tile load Global -> LDS.
//   data_size = 2 bytes (bf16), tile = tile0 x tile1 elements,
//   tensor row stride = stride0 elements,
//   LDS padding: 16 B inserted after every 64 B => LDS row stride 80 B (LDT=40)
// D# bit layout per CDNA5 ISA ch.8 (groups 0/1; groups 2/3 zero for 2D).
// ---------------------------------------------------------------------------
__device__ __forceinline__ void tdm_load_tile_bf16(uint32_t lds_addr,
                                                   const void* gaddr,
                                                   uint32_t stride0,
                                                   uint32_t tile0,
                                                   uint32_t tile1)
{
    const uint64_t ga = (uint64_t)(uintptr_t)gaddr;
    u32x4 g0;
    g0[0] = 1u;                                            // count=1, user D#
    g0[1] = lds_addr;                                      // LDS byte address
    g0[2] = (uint32_t)ga;                                  // global_addr[31:0]
    g0[3] = (uint32_t)((ga >> 32) & 0x01FFFFFFu)           // global_addr[56:32]
          | (2u << 30);                                    // type = 2 (image)
    i32x8 g1;
    g1[0] = (int)((1u << 16)        // data_size = 1 -> 2 bytes
                | (1u << 20)        // pad_enable
                | (3u << 22)        // pad_interval: 16 DWORDs (64 B)
                | (3u << 25));      // pad_amount:   4 DWORDs (16 B)
    const uint32_t tdim0 = 1u << 20;                       // large: no OOB clip
    const uint32_t tdim1 = 1u << 20;
    g1[1] = (int)((tdim0 & 0xFFFFu) << 16);                // tensor_dim0 lo
    g1[2] = (int)(((tdim0 >> 16) & 0xFFFFu) | ((tdim1 & 0xFFFFu) << 16));
    g1[3] = (int)(((tdim1 >> 16) & 0xFFFFu) | ((tile0 & 0xFFFFu) << 16));
    g1[4] = (int)(tile1 & 0xFFFFu);                        // tile_dim1, tile_dim2=0
    g1[5] = (int)stride0;                                  // tensor_dim0_stride lo
    g1[6] = 0;                                             // stride0 hi, dim1_stride lo
    g1[7] = 0;
    const i32x4 z4 = {0, 0, 0, 0};
#if TDM_6ARG
    const i32x8 z8 = {0, 0, 0, 0, 0, 0, 0, 0};
    __builtin_amdgcn_tensor_load_to_lds(g0, g1, z4, z4, z8, 0);
#else
    __builtin_amdgcn_tensor_load_to_lds(g0, g1, z4, z4, 0);
#endif
}

// Async 4-byte Global -> LDS load (per-lane LDS destination), ASYNCcnt tracked.
__device__ __forceinline__ void async_load_f32_to_lds(float* ldsDst,
                                                      const float* gsrc)
{
    uint32_t la = (uint32_t)(uintptr_t)ldsDst;
    uint64_t ga = (uint64_t)(uintptr_t)gsrc;
    asm volatile("global_load_async_to_lds_b32 %0, %1, off"
                 :: "v"(la), "v"(ga) : "memory");
}
__device__ __forceinline__ void wait_async0()
{
    asm volatile("s_wait_asynccnt 0x0" ::: "memory");
}

// ---------------------------------------------------------------------------
// Kernel 1: chaotic (logistic-map) + Padilha wave modulation of query,
//           constant scalar modulation of memory; outputs bf16 for WMMA.
// ---------------------------------------------------------------------------
__global__ __launch_bounds__(256)
void modulate_kernel(const float* __restrict__ query,
                     const float* __restrict__ memory,
                     bf16* __restrict__ qmod,
                     bf16* __restrict__ mmod)
{
    const int r   = blockIdx.x;        // row = b*SEQ + s
    const int tid = threadIdx.x;       // 256
    const int s   = r & (SEQ - 1);
    const float* qrow = query  + (size_t)r * D4;
    const float* mrow = memory + (size_t)r * D4;

    __shared__ float red[256];
    float ss = 0.f;
    for (int i = tid; i < D4; i += 256) { float v = qrow[i]; ss += v * v; }
    red[tid] = ss; __syncthreads();
    for (int off = 128; off > 0; off >>= 1) {
        if (tid < off) red[tid] += red[tid + off];
        __syncthreads();
    }
    const float nrm = sqrtf(red[0]);

    float x = 1.f / (1.f + __expf(-nrm));
#pragma unroll
    for (int it = 0; it < 10; ++it) x = 3.9f * x * (1.f - x);
    const float f1 = 1.f + 0.1f * (x - 0.5f);

    const float lam   = (float)s / (float)SEQ;
    float alpha = 1.0f * (1.f + 0.5f * (1.5f - 2.0f) / 2.0f);      // 0.875
    alpha = fminf(fmaxf(alpha, 0.1f), 3.0f);
    const float beta  = 2.f * 1.f + 1.f - 2.f * 1.5f;              // 0
    const float wave  = __sinf(alpha * lam) *
                        __cosf(-2.0f * lam + beta * lam * lam);
    const float fq = f1 * (1.f + 0.1f * wave);

    const float dop = 0.9f * 0.5f + 0.1f / (1.f + __expf(-0.7f));
    const float ser = 0.9f * 0.5f + 0.1f / (1.f + __expf(-0.8f));
    const float nor = 0.9f * 0.5f + 0.1f / (1.f + __expf(-0.6f));
    const float ms  = 0.4f * dop + 0.3f * ser + 0.3f * nor;

    for (int i = tid; i < D4; i += 256) {
        qmod[(size_t)r * D4 + i] = (bf16)(qrow[i] * fq);
        mmod[(size_t)r * D4 + i] = (bf16)(mrow[i] * ms);
    }
}

// ---------------------------------------------------------------------------
// Kernel 2: weight conversion f32 -> bf16
// ---------------------------------------------------------------------------
__global__ __launch_bounds__(256)
void cvt_bf16_kernel(const float* __restrict__ src, bf16* __restrict__ dst, int n)
{
    const int i = blockIdx.x * 256 + threadIdx.x;
    if (i < n) dst[i] = (bf16)src[i];
}

// ---------------------------------------------------------------------------
// Kernel 3: C[M,1024] = A[M,1024](bf16) @ W[1024,1024]^T (bf16) + bias
// 128x128 block tile, BK=32, 8 waves of 64x32 (4x2 WMMA tiles).
// Tile staging via Tensor Data Mover (double-buffered, overlapped with WMMA):
//   wave 0 issues the A-tile DMA, wave 1 the B-tile DMA, TENSORcnt-synced.
// W row-major [n][k] is exactly the column-major B matrix the fragment wants.
// ---------------------------------------------------------------------------
#define BM  128
#define BN  128
#define BK  32
#define LDT 40      // LDS row stride in elements (80 B) — matches TDM padding

__global__ __launch_bounds__(256)
void gemm_bf16_wmma_kernel(const bf16* __restrict__ A,
                           const bf16* __restrict__ W,
                           const float* __restrict__ bias,
                           float* __restrict__ C)
{
    __shared__ bf16 As[2][BM * LDT];
    __shared__ bf16 Bs[2][BN * LDT];

    const int tid  = threadIdx.x;
    const int lane = tid & 31;
    const int wave = tid >> 5;              // 0..7
    const int wm   = (wave >> 2) * 64;      // 0 / 64
    const int wn   = (wave & 3) * 32;       // 0,32,64,96
    const int lr   = lane & 15;             // fragment row
    const int kb   = (lane >> 4) * 8;       // fragment k-base (0 or 8)

    const int blockN = blockIdx.x * BN;
    const int blockM = blockIdx.y * BM;

    const bf16* aTile = A + (size_t)blockM * D4;   // tile row 0, k offset added per step
    const bf16* wTile = W + (size_t)blockN * D4;

    // per-thread prefetch addresses (L2 warm-up ahead of the TDM)
    const int r0   = tid >> 2;
    const int part = tid & 3;
    const bf16* aPf = A + (size_t)(blockM + r0) * D4 + part * 8;
    const bf16* wPf = W + (size_t)(blockN + r0) * D4 + part * 8;

    auto issue_tiles = [&](int kt, int buf) {
        const size_t ko = (size_t)kt * BK;
        if (wave == 0)
            tdm_load_tile_bf16((uint32_t)(uintptr_t)&As[buf][0],
                               aTile + ko, D4, BK, BM);
        else if (wave == 1)
            tdm_load_tile_bf16((uint32_t)(uintptr_t)&Bs[buf][0],
                               wTile + ko, D4, BK, BN);
    };

    v8f acc[4][2];
#pragma unroll
    for (int i = 0; i < 4; ++i)
#pragma unroll
        for (int j = 0; j < 2; ++j)
            acc[i][j] = (v8f){0.f, 0.f, 0.f, 0.f, 0.f, 0.f, 0.f, 0.f};

    const int NT = D4 / BK;                 // 32 k-steps

    issue_tiles(0, 0);
    if (wave < 2) __builtin_amdgcn_s_wait_tensorcnt(0);
    __syncthreads();

    for (int kt = 0; kt < NT; ++kt) {
        const int cur = kt & 1;
        if (kt + 1 < NT) issue_tiles(kt + 1, cur ^ 1);  // DMA overlaps WMMA
        if (kt + 2 < NT) {                               // L2 prefetch 2 ahead
            __builtin_prefetch(aPf + (size_t)(kt + 2) * BK, 0, 0);
            __builtin_prefetch(wPf + (size_t)(kt + 2) * BK, 0, 0);
        }

        union FragU { v16bf v; uint4 q[2]; };
        FragU a[4], b[2];
#pragma unroll
        for (int i = 0; i < 4; ++i) {
            const bf16* p = &As[cur][(wm + 16 * i + lr) * LDT];
            a[i].q[0] = *(const uint4*)(p + kb);
            a[i].q[1] = *(const uint4*)(p + 16 + kb);
        }
#pragma unroll
        for (int j = 0; j < 2; ++j) {
            const bf16* p = &Bs[cur][(wn + 16 * j + lr) * LDT];
            b[j].q[0] = *(const uint4*)(p + kb);
            b[j].q[1] = *(const uint4*)(p + 16 + kb);
        }
#pragma unroll
        for (int i = 0; i < 4; ++i)
#pragma unroll
            for (int j = 0; j < 2; ++j)
                acc[i][j] = __builtin_amdgcn_wmma_f32_16x16x32_bf16(
                    false, a[i].v, false, b[j].v, (short)0, acc[i][j],
                    false, false);

        if (kt + 1 < NT && wave < 2) __builtin_amdgcn_s_wait_tensorcnt(0);
        __syncthreads();
    }

    // epilogue: C layout — VGPR v holds M = v + 8*(lane/16), N = lane%16
    const int ml = 8 * (lane >> 4);
#pragma unroll
    for (int i = 0; i < 4; ++i) {
#pragma unroll
        for (int j = 0; j < 2; ++j) {
            const int col = blockN + wn + 16 * j + (lane & 15);
            const float bb = bias[col];
            float* crow = C + (size_t)(blockM + wm + 16 * i + ml) * D4 + col;
#pragma unroll
            for (int v = 0; v < 8; ++v)
                crow[(size_t)v * D4] = acc[i][j][v] + bb;
        }
    }
}

// ---------------------------------------------------------------------------
// Kernel 4: quaternionic spectral attention.
// One block per (batch b, quaternion lane j in [0,256)); components at columns
// j, j+256, j+512, j+768. 12 forward FFT(2048)+filter, 2 Hamilton products,
// 4 inverse FFT(2048) — in 128 KB dynamic LDS (WGP has 320 KB).
// Strided column loads use GLOBAL_LOAD_ASYNC_TO_LDS_B32 (ASYNCcnt tracked).
// ---------------------------------------------------------------------------
__device__ __forceinline__ float2 cmul(float2 a, float2 b) {
    return make_float2(a.x * b.x - a.y * b.y, a.x * b.y + a.y * b.x);
}
__device__ __forceinline__ float2 cadd(float2 a, float2 b) {
    return make_float2(a.x + b.x, a.y + b.y);
}
__device__ __forceinline__ float2 csub(float2 a, float2 b) {
    return make_float2(a.x - b.x, a.y - b.y);
}

__device__ __forceinline__ void fft_stages(float2* buf, int tid, float sign)
{
    // iterative radix-2 DIT; input bit-reversed, output natural order
#pragma unroll 1
    for (int st = 1; st <= 11; ++st) {
        const int half = 1 << (st - 1);
        const float ang = sign * (-6.283185307179586f) / (float)(1 << st);
        for (int i = tid; i < SEQ / 2; i += 256) {
            const int p    = i & (half - 1);
            const int base = ((i >> (st - 1)) << st) + p;
            float sw, cw;
            __sincosf(ang * (float)p, &sw, &cw);
            const float2 w = make_float2(cw, sw);
            const float2 u = buf[base];
            const float2 t = cmul(w, buf[base + half]);
            buf[base]        = make_float2(u.x + t.x, u.y + t.y);
            buf[base + half] = make_float2(u.x - t.x, u.y - t.y);
        }
        __syncthreads();
    }
}

__device__ __forceinline__ float2 spec_filter(int s) {
    const float th = 1.5f * atanf(__logf((float)s + 1e-10f));
    float sn, cs; __sincosf(th, &sn, &cs);
    return make_float2(cs, sn);
}

__device__ __forceinline__ void hamilton4(const float2 a[4], const float2 q[4],
                                          float2 o[4])
{
    o[0] = csub(csub(csub(cmul(a[0], q[0]), cmul(a[1], q[1])),
                     cmul(a[2], q[2])), cmul(a[3], q[3]));
    o[1] = csub(cadd(cadd(cmul(a[0], q[1]), cmul(a[1], q[0])),
                     cmul(a[2], q[3])), cmul(a[3], q[2]));
    o[2] = cadd(cadd(csub(cmul(a[0], q[2]), cmul(a[1], q[3])),
                     cmul(a[2], q[0])), cmul(a[3], q[1]));
    o[3] = cadd(csub(cadd(cmul(a[0], q[3]), cmul(a[1], q[2])),
                     cmul(a[2], q[1])), cmul(a[3], q[0]));
}

// load real column (stride D4 floats) bit-reversed into LDS complex buffer
__device__ __forceinline__ void load_col_fft_filt(float2* buf, const float* src,
                                                  int tid)
{
    for (int s = tid; s < SEQ; s += 256) {
        const int r = __brev((unsigned)s) >> 21;
        buf[r].y = 0.f;                                   // imag via DS store
        async_load_f32_to_lds(&buf[r].x, src + (size_t)s * D4);
    }
    wait_async0();
    __syncthreads();
    fft_stages(buf, tid, 1.f);
    for (int s = tid; s < SEQ; s += 256) buf[s] = cmul(buf[s], spec_filter(s));
    __syncthreads();
}

__global__ __launch_bounds__(256)
void spectral_kernel(const float* __restrict__ Qm, const float* __restrict__ Km,
                     const float* __restrict__ Vm, float* __restrict__ Out)
{
    extern __shared__ float2 sm[];
    float2* QK = sm;                 // 4 * 2048 complex
    float2* VF = sm + 4 * SEQ;       // 4 * 2048 complex
    const int tid = threadIdx.x;     // 256
    const int j   = blockIdx.x;      // 0..255
    const int b   = blockIdx.y;      // 0..3
    const size_t base = (size_t)b * SEQ * D4;

    for (int c = 0; c < 4; ++c)      // Qf * filt -> QK
        load_col_fft_filt(QK + c * SEQ, Qm + base + (j + 256 * c), tid);
    for (int c = 0; c < 4; ++c)      // Kf * filt -> VF
        load_col_fft_filt(VF + c * SEQ, Km + base + (j + 256 * c), tid);

    // QK = hamilton(Qf*filt, Kf*filt), pointwise in frequency
    for (int s = tid; s < SEQ; s += 256) {
        float2 a[4], q[4], o[4];
#pragma unroll
        for (int c = 0; c < 4; ++c) { a[c] = QK[c * SEQ + s]; q[c] = VF[c * SEQ + s]; }
        hamilton4(a, q, o);
#pragma unroll
        for (int c = 0; c < 4; ++c) QK[c * SEQ + s] = o[c];
    }
    __syncthreads();

    for (int c = 0; c < 4; ++c)      // Vf * filt -> VF (overwrites Kf)
        load_col_fft_filt(VF + c * SEQ, Vm + base + (j + 256 * c), tid);

    // QKV = hamilton(QK, Vf*filt), in place over QK
    for (int s = tid; s < SEQ; s += 256) {
        float2 a[4], q[4], o[4];
#pragma unroll
        for (int c = 0; c < 4; ++c) { a[c] = QK[c * SEQ + s]; q[c] = VF[c * SEQ + s]; }
        hamilton4(a, q, o);
#pragma unroll
        for (int c = 0; c < 4; ++c) QK[c * SEQ + s] = o[c];
    }
    __syncthreads();

    // inverse FFT per component, write real part
    const float invn = 1.f / (float)SEQ;
    for (int c = 0; c < 4; ++c) {
        float2* qb = QK + c * SEQ;
        float2* ib = VF + c * SEQ;
        for (int s = tid; s < SEQ; s += 256)
            ib[__brev((unsigned)s) >> 21] = qb[s];
        __syncthreads();
        fft_stages(ib, tid, -1.f);
        float* dst = Out + base + (j + 256 * c);
        for (int s = tid; s < SEQ; s += 256)
            dst[(size_t)s * D4] = ib[s].x * invn;
        __syncthreads();
    }
}

// ---------------------------------------------------------------------------
// launch
// ---------------------------------------------------------------------------
extern "C" void kernel_launch(void* const* d_in, const int* in_sizes, int n_in,
                              void* d_out, int out_size, void* d_ws, size_t ws_size,
                              hipStream_t stream)
{
    const float* query  = (const float*)d_in[0];
    const float* memory = (const float*)d_in[1];
    const float* Wq = (const float*)d_in[2];
    const float* bq = (const float*)d_in[3];
    const float* Wk = (const float*)d_in[4];
    const float* bk = (const float*)d_in[5];
    const float* Wv = (const float*)d_in[6];
    const float* bv = (const float*)d_in[7];
    float* out = (float*)d_out;

    char* ws = (char*)d_ws;
    bf16*  qmod = (bf16*)ws;                                   // 16 MB
    bf16*  mmod = (bf16*)(ws + (size_t)16 * 1024 * 1024);      // 16 MB
    bf16*  Wqb  = (bf16*)(ws + (size_t)32 * 1024 * 1024);      //  2 MB
    bf16*  Wkb  = (bf16*)(ws + (size_t)34 * 1024 * 1024);      //  2 MB
    bf16*  Wvb  = (bf16*)(ws + (size_t)36 * 1024 * 1024);      //  2 MB
    float* Kf   = (float*)(ws + (size_t)38 * 1024 * 1024);     // 32 MB
    float* Vf   = (float*)(ws + (size_t)70 * 1024 * 1024);     // 32 MB

    modulate_kernel<<<MROWS, 256, 0, stream>>>(query, memory, qmod, mmod);

    const int WN = D4 * D4;
    cvt_bf16_kernel<<<WN / 256, 256, 0, stream>>>(Wq, Wqb, WN);
    cvt_bf16_kernel<<<WN / 256, 256, 0, stream>>>(Wk, Wkb, WN);
    cvt_bf16_kernel<<<WN / 256, 256, 0, stream>>>(Wv, Wvb, WN);

    dim3 gg(D4 / BN, MROWS / BM);  // (8, 64)
    gemm_bf16_wmma_kernel<<<gg, 256, 0, stream>>>(qmod, Wqb, bq, out); // Q -> d_out
    gemm_bf16_wmma_kernel<<<gg, 256, 0, stream>>>(mmod, Wkb, bk, Kf);
    gemm_bf16_wmma_kernel<<<gg, 256, 0, stream>>>(mmod, Wvb, bv, Vf);

    // 128 KB dynamic LDS per block (WGP supports up to 320 KB)
    (void)hipFuncSetAttribute((const void*)spectral_kernel,
                              hipFuncAttributeMaxDynamicSharedMemorySize, 131072);
    spectral_kernel<<<dim3(256, BATCH), 256, 131072, stream>>>(out, Kf, Vf, out);
}
